// ActivationQuantizer_12687333392629
// MI455X (gfx1250) — compile-verified, matrix operand order checked by reading, not run on hardware
//
#include <hip/hip_runtime.h>
#include <cstdint>

// Problem constants (fixed by the reference).
#define NB     8
#define NCH    128
#define HW     56
#define PP     (HW*HW)        // 3136 spatial positions
#define NELEM  (NB*NCH*PP)    // 3,211,264
#define RROWS  (NCH*9)        // 1152
#define NCOLS  (PP*NB)        // 25088
#define K1IDX  231            // ceil(1152*0.2)
#define K2IDX  5018           // ceil(25088*0.2)

// Workspace layout (32-bit words)
#define W_KMIN 0
#define W_KMAX 1
#define W_R1   2
#define W_R2   3
#define W_NZR  16                       // 1152 ints
#define W_HIST 1200                     // 1153 ints (nzc values <= 1152)
#define W_T    2368                     // 25088 ints: T[b*3136+p] = sum_c [q!=0]
#define W_NZC  (W_T + NCOLS)            // 25088 ints
#define W_VAL  (W_NZC + NCOLS)          // 52544 (16B aligned); NELEM floats
#define W_NEED ((size_t)(W_VAL + NELEM) * 4)

typedef int v8i __attribute__((ext_vector_type(8)));

// Order-preserving float<->uint key (deterministic atomic min/max).
__device__ __forceinline__ unsigned fkey(float f) {
  unsigned u = __float_as_uint(f);
  return (u & 0x80000000u) ? ~u : (u | 0x80000000u);
}
__device__ __forceinline__ float fdecode(unsigned k) {
  unsigned u = (k & 0x80000000u) ? (k ^ 0x80000000u) : ~k;
  return __uint_as_float(u);
}
__device__ __forceinline__ float get_scale(const unsigned* W) {
  return (fdecode(W[W_KMAX]) - fdecode(W[W_KMIN])) / 255.0f;
}
// MODE==0: v is already scale*rint(x/scale). MODE==1: v is raw x.
template <int MODE>
__device__ __forceinline__ float finish(float v, float scale) {
  return MODE ? scale * rintf(v / scale) : v;
}
template <int MODE>
__device__ __forceinline__ int nztest(float v, float scale) {
  return MODE ? (rintf(v / scale) != 0.0f) : (v != 0.0f);
}

// ---------------- K0: init workspace ----------------
__global__ void init_ws(unsigned* W) {
  for (int i = threadIdx.x; i < W_T; i += blockDim.x) W[i] = 0u;
  if (threadIdx.x == 0) W[W_KMIN] = 0xFFFFFFFFu;   // +inf key
}

// ---------------- K1: global min/max ----------------
__global__ __launch_bounds__(256) void minmax_kernel(const float* __restrict__ x,
                                                     unsigned* W, int n4) {
  __shared__ float smn[256], smx[256];
  float mn = 3.402823466e38f, mx = -3.402823466e38f;
  const float4* x4 = (const float4*)x;
  for (int i = blockIdx.x * blockDim.x + threadIdx.x; i < n4;
       i += gridDim.x * blockDim.x) {
    float4 v = x4[i];
    mn = fminf(mn, fminf(fminf(v.x, v.y), fminf(v.z, v.w)));
    mx = fmaxf(mx, fmaxf(fmaxf(v.x, v.y), fmaxf(v.z, v.w)));
  }
  smn[threadIdx.x] = mn; smx[threadIdx.x] = mx;
  __syncthreads();
  for (int s = 128; s > 0; s >>= 1) {
    if (threadIdx.x < s) {
      smn[threadIdx.x] = fminf(smn[threadIdx.x], smn[threadIdx.x + s]);
      smx[threadIdx.x] = fmaxf(smx[threadIdx.x], smx[threadIdx.x + s]);
    }
    __syncthreads();
  }
  if (threadIdx.x == 0) {
    atomicMin(&W[W_KMIN], fkey(smn[0]));
    atomicMax(&W[W_KMAX], fkey(smx[0]));
  }
}

// ---------------- K1b: val[i] = scale*rint(x[i]/scale) (quantize ONCE) -------
__global__ __launch_bounds__(256) void prep_kernel(const float* __restrict__ x,
                                                   const unsigned* __restrict__ W,
                                                   float* __restrict__ val, int n4) {
  const float scale = get_scale(W);
  const float4* x4 = (const float4*)x;
  float4* v4 = (float4*)val;
  for (int i = blockIdx.x * blockDim.x + threadIdx.x; i < n4;
       i += gridDim.x * blockDim.x) {
    float4 v = x4[i];
    float4 o;
    o.x = scale * rintf(v.x / scale);
    o.y = scale * rintf(v.y / scale);
    o.z = scale * rintf(v.z / scale);
    o.w = scale * rintf(v.w / scale);
    v4[i] = o;
  }
}

// ---------------- K2: T[b,p] = sum_c [q != 0] via WMMA IU8 ----------------
// 8 waves/block, one 16-position tile per wave. A(16x64 u8) = nz indicators
// (M=pos, K=chan), B = all-ones -> D[m,*] = channel count for position m.
template <int MODE>
__global__ __launch_bounds__(256) void channel_count_kernel(const float* __restrict__ src,
                                                            const unsigned* __restrict__ W,
                                                            int* __restrict__ T) {
  const int tile  = blockIdx.x * 8 + (threadIdx.x >> 5);   // 0 .. 1567
  const int b     = tile / (PP / 16);
  const int pbase = (tile % (PP / 16)) * 16;
  const float scale = get_scale(W);
  const int lane = threadIdx.x & 31;     // wave32
  const int hi   = lane >> 4;            // lane group (K offset +8)
  const int m    = lane & 15;            // matrix row = position
  const float* plane = src + (size_t)b * NCH * PP + (pbase + m);

  v8i acc, ones;
#pragma unroll
  for (int j = 0; j < 8; ++j) { acc[j] = 0; ones[j] = 0x01010101; }

#pragma unroll
  for (int chunk = 0; chunk < 2; ++chunk) {
    const int cbase = chunk * 64 + hi * 8;
    v8i a;
#pragma unroll
    for (int j = 0; j < 8; ++j) {
      // ISA 8-bit A layout: VGPR j holds K = (j>>1)*16 + (j&1)*4 (+8 for hi lanes)
      const int kb = (j >> 1) * 16 + (j & 1) * 4 + cbase;
      unsigned pk = 0;
#pragma unroll
      for (int t = 0; t < 4; ++t) {
        float v = plane[(size_t)(kb + t) * PP];   // coalesced across the 16 lanes
        pk |= (unsigned)nztest<MODE>(v, scale) << (8 * t);
      }
      a[j] = (int)pk;
    }
    acc = __builtin_amdgcn_wmma_i32_16x16x64_iu8(false, a, false, ones, acc,
                                                 false, false);
  }
  // D layout: lane 0 holds (M=0..7, N=0) in dwords 0..7; lane 16 holds M=8..15.
  if (m == 0) {
    int* dst = T + b * PP + pbase + hi * 8;
#pragma unroll
    for (int j = 0; j < 8; ++j) dst[j] = acc[j];
  }
}

// ---------------- K3: nzc = 3x3 box-sum of T (+ histogram) ----------------
__global__ __launch_bounds__(256) void nzc_kernel(const int* __restrict__ T,
                                                  int* __restrict__ nzc,
                                                  int* __restrict__ hist) {
  int col = blockIdx.x * blockDim.x + threadIdx.x;
  if (col >= NCOLS) return;
  int b = col & 7, p = col >> 3;
  int oy = p / HW, ox = p % HW;
  const int* Tb = T + b * PP;
  int s = 0;
#pragma unroll
  for (int dy = -1; dy <= 1; ++dy) {
    int y = oy + dy;
    if ((unsigned)y >= HW) continue;
#pragma unroll
    for (int dx = -1; dx <= 1; ++dx) {
      int xx = ox + dx;
      if ((unsigned)xx >= HW) continue;
      s += Tb[y * HW + xx];
    }
  }
  nzc[col] = s;
  atomicAdd(&hist[s], 1);
}

// ---------------- K4: nzr via per-plane window sums ----------------
template <int MODE>
__global__ __launch_bounds__(64) void nzr_kernel(const float* __restrict__ src,
                                                 const unsigned* __restrict__ W,
                                                 int* __restrict__ nzr) {
  __shared__ int sF[HW], sNF[HW], sNL[HW];
  const float scale = get_scale(W);
  const int plane = blockIdx.x;          // b*128 + c
  const int c = plane % NCH;
  const int y = threadIdx.x;
  if (y < HW) {
    const float4* row = (const float4*)(src + (size_t)plane * PP + y * HW);
    int full = 0, nz0 = 0, nz55 = 0;
#pragma unroll
    for (int xx = 0; xx < HW / 4; ++xx) {
      float4 v = row[xx];
      int n0 = nztest<MODE>(v.x, scale), n1 = nztest<MODE>(v.y, scale);
      int n2 = nztest<MODE>(v.z, scale), n3 = nztest<MODE>(v.w, scale);
      full += n0 + n1 + n2 + n3;
      if (xx == 0)          nz0  = n0;
      if (xx == HW / 4 - 1) nz55 = n3;
    }
    sF[y]  = full;
    sNF[y] = full - nz0;     // exclude first col (fj==2)
    sNL[y] = full - nz55;    // exclude last col  (fj==0)
  }
  __syncthreads();
  const int t = threadIdx.x;
  if (t < 9) {
    int fi = t / 3, fj = t % 3;
    int y0 = (fi == 2) ? 1 : 0;
    int y1 = (fi == 0) ? (HW - 2) : (HW - 1);
    int sum = 0;
    for (int yy = y0; yy <= y1; ++yy)
      sum += (fj == 0) ? sNL[yy] : (fj == 1) ? sF[yy] : sNF[yy];
    atomicAdd(&nzr[c * 9 + t], sum);
  }
}

// ---------------- K5: r1 = sorted(nzr)[231] via rank counting ----------------
__global__ __launch_bounds__(256) void select_r1(const int* __restrict__ nzr,
                                                 int* __restrict__ r1) {
  int i = blockIdx.x * blockDim.x + threadIdx.x;
  if (i >= RROWS) return;
  int v = nzr[i], less = 0, eq = 0;
  for (int j = 0; j < RROWS; ++j) {
    int u = nzr[j];
    less += (u < v);
    eq   += (u == v);
  }
  if (less <= K1IDX && K1IDX < less + eq) *r1 = v;  // all writers write same v
}

// ---------------- K6: r2 = sorted(nzc)[5018] via histogram CDF ----------------
__global__ __launch_bounds__(256) void select_r2(const int* __restrict__ hist,
                                                 int* __restrict__ r2) {
  int v = blockIdx.x * blockDim.x + threadIdx.x;
  if (v > RROWS) return;                 // bins 0..1152
  int h = hist[v];
  if (h == 0) return;
  int cdf = 0;
  for (int u = 0; u < v; ++u) cdf += hist[u];
  if (cdf <= K2IDX && K2IDX < cdf + h) *r2 = v;
}

// ---------------- K7: output (async-to-LDS staged im2col, 9x reuse) ----------
// Block = (channel c, 7-row band oy0..oy0+6). LDS tile holds src values for all
// 8 batches, rows y in [oy0-1, oy0+7], x in [-1,56] (zero padded), skew per b.
#define TY       7
#define ROWSTR   64                      // LDS floats per (b,yy) row; x at idx x+4
#define BSTR     (9 * ROWSTR + 4)        // 580: +4 skew kills bank conflicts
#define TILEF    (NB * BSTR)             // 4640 floats

template <int MODE>
__global__ __launch_bounds__(256) void output_kernel(const float* __restrict__ src,
                                                     const unsigned* __restrict__ W,
                                                     const int* __restrict__ nzr,
                                                     const int* __restrict__ nzc,
                                                     const int* __restrict__ r1p,
                                                     const int* __restrict__ r2p,
                                                     float* __restrict__ out) {
  __shared__ float tile[TILEF];
  __shared__ unsigned okc4[TY * HW * NB / 4];   // 784 words: per-col mask bytes
  const int c   = blockIdx.x >> 3;
  const int oy0 = (blockIdx.x & 7) * TY;
  const int tid = threadIdx.x;
  const float scale = get_scale(W);
  const int r1 = *r1p, r2 = *r2p;

  // Zero the tile (covers padding and out-of-range rows).
  for (int i = tid; i < TILEF; i += 256) tile[i] = 0.0f;
  __syncthreads();

  // Async-copy interior: 8 batches x 9 rows x 14 x 16B chunks (16B aligned).
  for (int job = tid; job < NB * 9 * 14; job += 256) {
    int chunk = job % 14;
    int tmp   = job / 14;
    int yy    = tmp % 9;
    int b     = tmp / 9;
    int y     = oy0 - 1 + yy;
    if (y >= 0 && y < HW) {
      uint64_t gaddr = (uint64_t)(src + ((size_t)(b * NCH + c) * HW + y) * HW + chunk * 4);
      unsigned laddr = (unsigned)(size_t)&tile[b * BSTR + yy * ROWSTR + 4 + chunk * 4];
      asm volatile("global_load_async_to_lds_b128 %0, %1, off"
                   :: "v"(laddr), "v"(gaddr) : "memory");
    }
  }
  // Per-column mask for this band (overlaps with async copies in flight).
  const int* ncrow = nzc + oy0 * HW * NB;
  for (int k = tid; k < TY * HW * NB / 4; k += 256) {
    int4 nv = ((const int4*)ncrow)[k];
    unsigned w = (nv.x >= r2 ? 1u : 0u) | (nv.y >= r2 ? 0x100u : 0u) |
                 (nv.z >= r2 ? 0x10000u : 0u) | (nv.w >= r2 ? 0x1000000u : 0u);
    okc4[k] = w;
  }
  asm volatile("s_wait_asynccnt 0" ::: "memory");
  __syncthreads();

  // Emit the 9 output rows for this channel from the staged tile.
#pragma unroll 1
  for (int rr = 0; rr < 9; ++rr) {
    const int fi = rr / 3, fj = rr % 3;
    const int r  = c * 9 + rr;
    const bool rowOk = nzr[r] >= r1;
    float4* orow4 = (float4*)(out + (size_t)r * NCOLS + oy0 * HW * NB);
    for (int k = tid; k < TY * HW * NB / 4; k += 256) {   // coalesced b128 stores
      int pl = k >> 1;                 // spatial position (local)
      int b0 = (k & 1) * 4;            // batches b0..b0+3
      int oyl = pl / HW, ox = pl % HW;
      int lbase = (oyl + fi) * ROWSTR + (ox + fj + 3);
      unsigned okw = okc4[k];
      float v0 = tile[(b0 + 0) * BSTR + lbase];
      float v1 = tile[(b0 + 1) * BSTR + lbase];
      float v2 = tile[(b0 + 2) * BSTR + lbase];
      float v3 = tile[(b0 + 3) * BSTR + lbase];
      float4 o;
      o.x = (rowOk && (okw & 0x1u))        ? finish<MODE>(v0, scale) : 0.0f;
      o.y = (rowOk && (okw & 0x100u))      ? finish<MODE>(v1, scale) : 0.0f;
      o.z = (rowOk && (okw & 0x10000u))    ? finish<MODE>(v2, scale) : 0.0f;
      o.w = (rowOk && (okw & 0x1000000u))  ? finish<MODE>(v3, scale) : 0.0f;
      orow4[k] = o;
    }
  }
}

extern "C" void kernel_launch(void* const* d_in, const int* in_sizes, int n_in,
                              void* d_out, int out_size, void* d_ws, size_t ws_size,
                              hipStream_t stream) {
  const float* x = (const float*)d_in[0];
  unsigned* W  = (unsigned*)d_ws;
  int*      Wi = (int*)d_ws;
  float*    out = (float*)d_out;

  const bool big = (ws_size >= W_NEED);   // true: precompute val; false: inline
  float* val = (float*)d_ws + W_VAL;
  const float* src = big ? (const float*)val : x;

  init_ws<<<1, 256, 0, stream>>>(W);
  minmax_kernel<<<512, 256, 0, stream>>>(x, W, NELEM / 4);
  if (big) {
    prep_kernel<<<1024, 256, 0, stream>>>(x, W, val, NELEM / 4);
    channel_count_kernel<0><<<NB * (PP / 16) / 8, 256, 0, stream>>>(src, W, Wi + W_T);
    nzr_kernel<0><<<NB * NCH, 64, 0, stream>>>(src, W, Wi + W_NZR);
  } else {
    channel_count_kernel<1><<<NB * (PP / 16) / 8, 256, 0, stream>>>(src, W, Wi + W_T);
    nzr_kernel<1><<<NB * NCH, 64, 0, stream>>>(src, W, Wi + W_NZR);
  }
  nzc_kernel<<<(NCOLS + 255) / 256, 256, 0, stream>>>(Wi + W_T, Wi + W_NZC, Wi + W_HIST);
  select_r1<<<(RROWS + 255) / 256, 256, 0, stream>>>(Wi + W_NZR, Wi + W_R1);
  select_r2<<<(RROWS + 1 + 255) / 256, 256, 0, stream>>>(Wi + W_HIST, Wi + W_R2);
  if (big)
    output_kernel<0><<<NCH * 8, 256, 0, stream>>>(src, W, Wi + W_NZR, Wi + W_NZC,
                                                  Wi + W_R1, Wi + W_R2, out);
  else
    output_kernel<1><<<NCH * 8, 256, 0, stream>>>(src, W, Wi + W_NZR, Wi + W_NZC,
                                                  Wi + W_R1, Wi + W_R2, out);
}
